// SBLocalAttention_47304769798175
// MI455X (gfx1250) — matrix operand order repaired and checked by reading, not compiled, not run on hardware
//
#include <hip/hip_runtime.h>
#include <math.h>
#include <stdint.h>

// ---- CDNA5 WMMA types ----
typedef __attribute__((ext_vector_type(16))) _Float16 v16h;
typedef __attribute__((ext_vector_type(8)))  float    v8f;
typedef __attribute__((ext_vector_type(4)))  unsigned int u32x4;
typedef __attribute__((ext_vector_type(8)))  int      i32x8;
typedef __attribute__((ext_vector_type(4)))  int      i32x4;

#define WMMA_F16(a, b, c) \
  __builtin_amdgcn_wmma_f32_16x16x32_f16(false, (a), false, (b), (short)0, (c), false, false)

// ---- problem constants (match reference) ----
constexpr int B_  = 2;
constexpr int T_  = 1024;
constexpr int H_  = 8;
constexpr int E_  = 64;
constexpr int M_  = 128;
constexpr int W_  = 64;
constexpr int BH_ = B_ * H_;

constexpr float TEMP       = 0.125f;                 // 1/sqrt(E)
constexpr float NORMALIZER = 0.35355339059327373f;   // sqrt(TEMP)
constexpr float MRSQRT     = 0.08838834764831845f;   // M^-0.5
constexpr float NEGBIG     = -1e24f;

// Fragment coordinate maps (wave32, 16x16x32 f16 WMMA):
//  A (16xK32): row = lane&15 ; K = e + 16*(lane>>4) (same order as B)   (e = half index 0..15)
//  B (K32x16): col = lane&15 ; K = e + 16*(lane>>4)
//  C (16x16) : row = r + 8*(lane>>4) ; col = lane&15                 (r = 0..7)

// ---------------------------------------------------------------------
// Tensor Data Mover: 2D f32 tile (tile_d1 rows x tile_d0 elems, row
// stride stride0 elems) from global -> LDS. D# built per ISA 8.3/8.4.
// ---------------------------------------------------------------------
__device__ __forceinline__ void tdm_load_2d_f32(
    unsigned lds_addr, const void* gptr,
    unsigned tile_d0, unsigned tile_d1,
    unsigned tensor_d0, unsigned tensor_d1,
    unsigned long long stride0_elems)
{
  unsigned long long ga = (unsigned long long)(uintptr_t)gptr;
  u32x4 g0;
  g0[0] = 1u;                                        // count=1 (valid), user mode
  g0[1] = lds_addr;                                  // LDS byte address
  g0[2] = (unsigned)(ga & 0xFFFFFFFFu);              // global_addr[31:0]
  g0[3] = (unsigned)((ga >> 32) & 0x01FFFFFFu)       // global_addr[56:32]
        | (2u << 30);                                // type=2 ("image")
  i32x8 g1;
  g1[0] = (int)(2u << 16);                           // data_size=2 (4 bytes)
  g1[1] = (int)((tensor_d0 & 0xFFFFu) << 16);        // tensor_dim0[15:0] @ [63:48]
  g1[2] = (int)((tensor_d0 >> 16) |                  // tensor_dim0[31:16]
                ((tensor_d1 & 0xFFFFu) << 16));      // tensor_dim1[15:0]
  g1[3] = (int)((tensor_d1 >> 16) |                  // tensor_dim1[31:16]
                ((tile_d0 & 0xFFFFu) << 16));        // tile_dim0
  g1[4] = (int)(tile_d1 & 0xFFFFu);                  // tile_dim1 ; tile_dim2=0
  g1[5] = (int)(stride0_elems & 0xFFFFFFFFull);      // tensor_dim0_stride[31:0]
  g1[6] = (int)((stride0_elems >> 32) & 0xFFFFull);  // stride0[47:32] ; stride1=0
  g1[7] = 0;
  i32x4 z4 = {0, 0, 0, 0};
#if __has_include(<hip/amd_detail/amd_gfx1250_TDM.h>)
  i32x8 z8 = {0, 0, 0, 0, 0, 0, 0, 0};
  __builtin_amdgcn_tensor_load_to_lds(g0, g1, z4, z4, z8, 0);
#else
  __builtin_amdgcn_tensor_load_to_lds(g0, g1, z4, z4, 0);
#endif
}

// =====================================================================
// Kernel 1: global max of k_log over (T, M) per head -> stab_k
// =====================================================================
__global__ __launch_bounds__(256) void k_stab_kernel(
    const float* __restrict__ key, const float* __restrict__ proj,
    float* __restrict__ stabk)
{
  __shared__ float diagk[T_];
  __shared__ float red[256];
  const int bh = blockIdx.x;
  const int b = bh / H_, h = bh % H_;
  const int tid = threadIdx.x;
  const int lane = tid & 31, wv = tid >> 5;
  const int laneLo = lane & 15, laneHi = lane >> 4;

  const float* kbase = key + ((size_t)b * T_ * H_ + h) * E_;

  for (int t = tid; t < T_; t += 256) {
    const float* kr = kbase + (size_t)t * H_ * E_;
    float s = 0.f;
    for (int e = 0; e < E_; ++e) { float x = kr[e]; s += x * x; }
    diagk[t] = 0.5f * TEMP * s;
  }
  __syncthreads();

  // hoist projection B-fragments (reused for all 64 row tiles)
  v16h bp[8][2];
  for (int mt = 0; mt < 8; ++mt)
    for (int ch = 0; ch < 2; ++ch) {
      v16h bb;
      for (int e = 0; e < 16; ++e) {
        int kk  = e + (laneHi << 4);
        int col = laneLo;
        bb[e] = (_Float16)proj[(mt * 16 + col) * E_ + ch * 32 + kk];
      }
      bp[mt][ch] = bb;
    }

  float vmax = -INFINITY;
  for (int rt = wv; rt < T_ / 16; rt += 8) {
    const int t0 = rt * 16;
    v16h a0, a1;
    for (int e = 0; e < 16; ++e) {
      int row = laneLo;
      int kk  = e + (laneHi << 4);
      const float* kr = kbase + (size_t)(t0 + row) * H_ * E_;
      a0[e] = (_Float16)(NORMALIZER * kr[kk]);
      a1[e] = (_Float16)(NORMALIZER * kr[32 + kk]);
    }
    for (int mt = 0; mt < 8; ++mt) {
      v8f c = {};
      c = WMMA_F16(a0, bp[mt][0], c);
      c = WMMA_F16(a1, bp[mt][1], c);
      for (int r = 0; r < 8; ++r) {
        int row = r + laneHi * 8;
        vmax = fmaxf(vmax, c[r] - diagk[t0 + row]);
      }
    }
  }
  red[tid] = vmax;
  __syncthreads();
  if (tid == 0) {
    float m = -INFINITY;
    for (int i = 0; i < 256; ++i) m = fmaxf(m, red[i]);
    stabk[bh] = m;
  }
}

// =====================================================================
// Kernel 2: k_prime (f16, workspace), kv = k_prime^T V  ([m][e]), ksum.
// Phase 2 stages V chunks into LDS with the Tensor Data Mover
// (double-buffered, wave 0 drives the TDM, TENSORcnt-synchronized).
// =====================================================================
__global__ __launch_bounds__(256) void k_feat_kernel(
    const float* __restrict__ key, const float* __restrict__ value,
    const float* __restrict__ proj, const float* __restrict__ stabk,
    _Float16* __restrict__ kprime, float* __restrict__ kv,
    float* __restrict__ ksum)
{
  __shared__ float diagk[T_];
  __shared__ float vtile[2][32][E_];   // 16 KB double buffer for TDM
  const int bh = blockIdx.x;
  const int b = bh / H_, h = bh % H_;
  const int tid = threadIdx.x;
  const int lane = tid & 31, wv = tid >> 5;
  const int laneLo = lane & 15, laneHi = lane >> 4;

  const float* kbase = key   + ((size_t)b * T_ * H_ + h) * E_;
  const float* vbase = value + ((size_t)b * T_ * H_ + h) * E_;
  const float stab = stabk[bh];

  for (int t = tid; t < T_; t += 256) {
    const float* kr = kbase + (size_t)t * H_ * E_;
    float s = 0.f;
    for (int e = 0; e < E_; ++e) { float x = kr[e]; s += x * x; }
    diagk[t] = 0.5f * TEMP * s;
  }
  __syncthreads();

  v16h bp[8][2];
  for (int mt0 = 0; mt0 < 8; ++mt0)
    for (int ch = 0; ch < 2; ++ch) {
      v16h bb;
      for (int e = 0; e < 16; ++e) {
        int kk  = e + (laneHi << 4);
        int col = laneLo;
        bb[e] = (_Float16)proj[(mt0 * 16 + col) * E_ + ch * 32 + kk];
      }
      bp[mt0][ch] = bb;
    }

  // phase 1: k_prime
  for (int rt = wv; rt < T_ / 16; rt += 8) {
    const int t0 = rt * 16;
    v16h a0, a1;
    for (int e = 0; e < 16; ++e) {
      int row = laneLo;
      int kk  = e + (laneHi << 4);
      const float* kr = kbase + (size_t)(t0 + row) * H_ * E_;
      a0[e] = (_Float16)(NORMALIZER * kr[kk]);
      a1[e] = (_Float16)(NORMALIZER * kr[32 + kk]);
    }
    for (int mt = 0; mt < 8; ++mt) {
      v8f c = {};
      c = WMMA_F16(a0, bp[mt][0], c);
      c = WMMA_F16(a1, bp[mt][1], c);
      for (int r = 0; r < 8; ++r) {
        int row = r + laneHi * 8;
        int t   = t0 + row;
        float kp = MRSQRT * expf((c[r] - diagk[t]) - stab);
        kprime[((size_t)bh * T_ + t) * M_ + mt * 16 + laneLo] = (_Float16)kp;
      }
    }
  }
  __threadfence();
  __syncthreads();

  // phase 2: kv[m][e] = sum_s kp[s][m] * v[s][e]
  // V chunk (32 x 64 f32, row stride H*E) arrives via TENSOR_LOAD_TO_LDS.
  const int mt = wv;  // 8 waves -> 8 m-tiles
  v8f ckv[4] = {{}, {}, {}, {}};

  if (wv == 0) {
    tdm_load_2d_f32((unsigned)(uintptr_t)&vtile[0][0][0], vbase,
                    E_, 32, E_, 32, (unsigned long long)(H_ * E_));
  }

  for (int sc = 0; sc < T_ / 32; ++sc) {
    const int s0 = sc * 32;
    if (wv == 0) __builtin_amdgcn_s_wait_tensorcnt(0);   // vtile[sc&1] ready
    __syncthreads();
    if (wv == 0 && sc + 1 < T_ / 32) {                   // prefetch next chunk
      tdm_load_2d_f32((unsigned)(uintptr_t)&vtile[(sc + 1) & 1][0][0],
                      vbase + (size_t)(s0 + 32) * H_ * E_,
                      E_, 32, E_, 32, (unsigned long long)(H_ * E_));
    }
    v16h a;
    for (int e = 0; e < 16; ++e) {
      int row = laneLo;                                  // m within tile
      int kk  = e + (laneHi << 4);                       // s within chunk
      a[e] = kprime[((size_t)bh * T_ + s0 + kk) * M_ + mt * 16 + row];
    }
    for (int et = 0; et < 4; ++et) {
      v16h bb;
      for (int e = 0; e < 16; ++e) {
        int kk  = e + (laneHi << 4);
        int col = laneLo;
        bb[e] = (_Float16)vtile[sc & 1][kk][et * 16 + col];
      }
      ckv[et] = WMMA_F16(a, bb, ckv[et]);
    }
  }
  for (int et = 0; et < 4; ++et)
    for (int r = 0; r < 8; ++r) {
      int m = mt * 16 + r + laneHi * 8;
      kv[((size_t)bh * M_ + m) * E_ + et * 16 + laneLo] = ckv[et][r];
    }

  if (tid < M_) {
    float s = 0.f;
    for (int t = 0; t < T_; ++t)
      s += (float)kprime[((size_t)bh * T_ + t) * M_ + tid];
    ksum[bh * M_ + tid] = s;
  }
}

// =====================================================================
// Kernel 3: per 16-row q tile — q features, qkv, local band, final fuse
// =====================================================================
__global__ __launch_bounds__(64) void out_kernel(
    const float* __restrict__ q, const float* __restrict__ k,
    const float* __restrict__ v, const float* __restrict__ proj,
    const float* __restrict__ stabk, const _Float16* __restrict__ kprime,
    const float* __restrict__ kv, const float* __restrict__ ksum,
    float* __restrict__ out)
{
  __shared__ float    qlog[2][16][M_];     // 16 KB
  __shared__ _Float16 qprime[2][16][M_];   //  8 KB
  __shared__ float    qkbuf[2][16][96];    // 12 KB (QK, later dots)
  __shared__ float    dpbuf[2][16][96];    // 12 KB
  __shared__ float    rowdiag[2][16], rowstab[2][16], rowqk1[2][16];
  __shared__ float    rowlognorm[2][16], rowpscale[2][16];

  const int bh = blockIdx.x;
  const int b = bh / H_, h = bh % H_;
  const int tid = threadIdx.x, wv = tid >> 5, lane = tid & 31;
  const int laneLo = lane & 15, laneHi = lane >> 4;
  const int t0 = (blockIdx.y * 2 + wv) * 16;

  const float* qbase = q + ((size_t)b * T_ * H_ + h) * E_;
  const float* kbase = k + ((size_t)b * T_ * H_ + h) * E_;
  const float* vbase = v + ((size_t)b * T_ * H_ + h) * E_;
  const float stab_k = stabk[bh];

  // warm the caches for the local window (global_prefetch_b8)
  {
    int sp = min(max(t0 - 32 + lane * 3, 0), T_ - 1);
    __builtin_prefetch(kbase + (size_t)sp * H_ * E_, 0, 0);
    __builtin_prefetch(vbase + (size_t)sp * H_ * E_, 0, 0);
    __builtin_prefetch(kprime + ((size_t)bh * T_ + sp) * M_, 0, 0);
  }

  if (lane < 16) {
    const float* qr = qbase + (size_t)(t0 + lane) * H_ * E_;
    float s = 0.f;
    for (int e = 0; e < E_; ++e) s += qr[e] * qr[e];
    rowdiag[wv][lane] = 0.5f * TEMP * s;
  }

  // normalized Q A-fragments (reused for q-features AND local QK)
  v16h aq0, aq1;
  for (int e = 0; e < 16; ++e) {
    int row = laneLo;
    int kk  = e + (laneHi << 4);
    const float* qr = qbase + (size_t)(t0 + row) * H_ * E_;
    aq0[e] = (_Float16)(NORMALIZER * qr[kk]);
    aq1[e] = (_Float16)(NORMALIZER * qr[32 + kk]);
  }
  __syncthreads();

  // q_log = (norm*Q)@proj^T - diag
  for (int mt = 0; mt < 8; ++mt) {
    v16h b0, b1;
    for (int e = 0; e < 16; ++e) {
      int kk  = e + (laneHi << 4);
      int col = laneLo;
      b0[e] = (_Float16)proj[(mt * 16 + col) * E_ + kk];
      b1[e] = (_Float16)proj[(mt * 16 + col) * E_ + 32 + kk];
    }
    v8f c = {};
    c = WMMA_F16(aq0, b0, c);
    c = WMMA_F16(aq1, b1, c);
    for (int r = 0; r < 8; ++r) {
      int row = r + 8 * laneHi;
      qlog[wv][row][mt * 16 + laneLo] = c[r] - rowdiag[wv][row];
    }
  }
  __syncthreads();

  if (lane < 16) {
    float m = -INFINITY;
    for (int mm = 0; mm < M_; ++mm) m = fmaxf(m, qlog[wv][lane][mm]);
    rowstab[wv][lane] = m;
  }
  __syncthreads();

  for (int idx = lane; idx < 16 * M_; idx += 32) {
    int r = idx >> 7, mm = idx & (M_ - 1);
    qprime[wv][r][mm] = (_Float16)(MRSQRT * expf(qlog[wv][r][mm] - rowstab[wv][r]));
  }
  __syncthreads();

  if (lane < 16) {
    float s = 0.f;
    for (int mm = 0; mm < M_; ++mm)
      s += (float)qprime[wv][lane][mm] * ksum[bh * M_ + mm];
    rowqk1[wv][lane] = s;
  }

  // q_prime A-fragments (4 chunks of K=32 over M)
  v16h aqp[4];
  for (int mc = 0; mc < 4; ++mc)
    for (int e = 0; e < 16; ++e) {
      int row = laneLo;
      int kk  = e + (laneHi << 4);
      aqp[mc][e] = qprime[wv][row][mc * 32 + kk];
    }

  // qkv = q_prime @ kv
  v8f cqkv[4] = {{}, {}, {}, {}};
  for (int et = 0; et < 4; ++et)
    for (int mc = 0; mc < 4; ++mc) {
      v16h bb;
      for (int e = 0; e < 16; ++e) {
        int kk  = e + (laneHi << 4);
        int col = laneLo;
        bb[e] = (_Float16)kv[((size_t)bh * M_ + mc * 32 + kk) * E_ + et * 16 + col];
      }
      cqkv[et] = WMMA_F16(aqp[mc], bb, cqkv[et]);
    }

  // local band: 96 columns covering s in [t0-32, t0+64)
  const int s0 = t0 - 32;

  // QK (normalizer on both sides => factor TEMP)
  for (int ct = 0; ct < 6; ++ct) {
    v16h b0, b1;
    for (int e = 0; e < 16; ++e) {
      int kk  = e + (laneHi << 4);
      int col = laneLo;
      int s   = s0 + ct * 16 + col;
      int scl = min(max(s, 0), T_ - 1);
      const float* kr = kbase + (size_t)scl * H_ * E_;
      b0[e] = (_Float16)(NORMALIZER * kr[kk]);
      b1[e] = (_Float16)(NORMALIZER * kr[32 + kk]);
    }
    v8f c = {};
    c = WMMA_F16(aq0, b0, c);
    c = WMMA_F16(aq1, b1, c);
    for (int r = 0; r < 8; ++r) {
      int row = r + 8 * laneHi;
      int cc  = ct * 16 + laneLo;
      int s   = s0 + cc;
      int w   = cc - row;
      bool valid = (w >= 0) && (w < W_) && (s >= 0) && (s < T_);
      qkbuf[wv][row][cc] = valid ? c[r] : NEGBIG;
    }
  }

  // dots_prime = q_prime · kp_win
  for (int ct = 0; ct < 6; ++ct) {
    v8f c = {};
    for (int mc = 0; mc < 4; ++mc) {
      v16h bb;
      for (int e = 0; e < 16; ++e) {
        int kk  = e + (laneHi << 4);
        int col = laneLo;
        int s   = s0 + ct * 16 + col;
        int scl = min(max(s, 0), T_ - 1);
        bb[e] = kprime[((size_t)bh * T_ + scl) * M_ + mc * 32 + kk];
      }
      c = WMMA_F16(aqp[mc], bb, c);
    }
    for (int r = 0; r < 8; ++r) {
      int row = r + 8 * laneHi;
      int cc  = ct * 16 + laneLo;
      int s   = s0 + cc;
      int w   = cc - row;
      bool valid = (w >= 0) && (w < W_) && (s >= 0) && (s < T_);
      dpbuf[wv][row][cc] = valid ? c[r] : 0.f;
    }
  }
  __syncthreads();

  // per-row log-space normalization
  if (lane < 16) {
    int r = lane;
    float mx = -INFINITY;
    for (int cc = 0; cc < 96; ++cc) mx = fmaxf(mx, qkbuf[wv][r][cc]);
    float se = 0.f;
    for (int cc = 0; cc < 96; ++cc) se += expf(qkbuf[wv][r][cc] - mx);
    float lse = mx + logf(se);
    float dps = 0.f;
    for (int cc = 0; cc < 96; ++cc) dps += dpbuf[wv][r][cc];
    float pls = rowstab[wv][r] + stab_k;
    float lr  = logf(fmaxf(rowqk1[wv][r] - dps, 1e-24f)) + pls;
    float a1 = fmaxf(lse, lr), a2 = fminf(lse, lr);
    float ln = a1 + log1pf(expf(a2 - a1));
    rowlognorm[wv][r] = ln;
    rowpscale[wv][r]  = expf(pls - ln);
  }
  __syncthreads();

  // dots (overwrite qkbuf)
  for (int idx = lane; idx < 16 * 96; idx += 32) {
    int r = idx / 96, cc = idx % 96;
    float d = expf(qkbuf[wv][r][cc] - rowlognorm[wv][r])
            - dpbuf[wv][r][cc] * rowpscale[wv][r];
    qkbuf[wv][r][cc] = d;
  }
  __syncthreads();

  // out_local = dots @ V_win ; fuse with qkv * prime_scale
  v16h ad[3];
  for (int kc = 0; kc < 3; ++kc)
    for (int e = 0; e < 16; ++e) {
      int row = laneLo;
      int kk  = e + (laneHi << 4);
      ad[kc][e] = (_Float16)qkbuf[wv][row][kc * 32 + kk];
    }

  for (int et = 0; et < 4; ++et) {
    v8f c = {};
    for (int kc = 0; kc < 3; ++kc) {
      v16h bb;
      for (int e = 0; e < 16; ++e) {
        int kk  = e + (laneHi << 4);
        int col = laneLo;
        int cc  = kc * 32 + kk;
        int s   = s0 + cc;
        int scl = min(max(s, 0), T_ - 1);
        bb[e] = (_Float16)vbase[(size_t)scl * H_ * E_ + et * 16 + col];
      }
      c = WMMA_F16(ad[kc], bb, c);
    }
    for (int r = 0; r < 8; ++r) {
      int row = r + 8 * laneHi;
      int t   = t0 + row;
      float val = c[r] + cqkv[et][r] * rowpscale[wv][row];
      out[((size_t)b * T_ + t) * H_ * E_ + (size_t)h * E_ + et * 16 + laneLo] = val;
    }
  }
}

// =====================================================================
extern "C" void kernel_launch(void* const* d_in, const int* in_sizes, int n_in,
                              void* d_out, int out_size, void* d_ws, size_t ws_size,
                              hipStream_t stream) {
  (void)in_sizes; (void)n_in; (void)out_size; (void)ws_size;
  const float* q    = (const float*)d_in[0];
  const float* k    = (const float*)d_in[1];
  const float* v    = (const float*)d_in[2];
  const float* proj = (const float*)d_in[3];
  float* out = (float*)d_out;

  // workspace: kprime f16 [BH][T][M] | stabk [BH] | kv [BH][M][E] | ksum [BH][M]
  _Float16* kprime = (_Float16*)d_ws;
  size_t off = (size_t)BH_ * T_ * M_ * sizeof(_Float16);
  float* stabk = (float*)((char*)d_ws + off);
  float* kv    = stabk + BH_;
  float* ksum  = kv + (size_t)BH_ * M_ * E_;

  k_stab_kernel<<<BH_, 256, 0, stream>>>(k, proj, stabk);
  k_feat_kernel<<<BH_, 256, 0, stream>>>(k, v, proj, stabk, kprime, kv, ksum);
  out_kernel<<<dim3(BH_, T_ / 32), 64, 0, stream>>>(q, k, v, proj, stabk,
                                                    kprime, kv, ksum, out);
}
